// p_A_x_func_8693013807337
// MI455X (gfx1250) — matrix operand order repaired and checked by reading, not compiled, not run on hardware
//
#include <hip/hip_runtime.h>
#include <hip/hip_bf16.h>
#include <math.h>

// Problem sizes (from reference)
#define BDIM   512     // batch
#define DDIM   512     // features
#define TFLEN  195     // valid rows of A (rest of K contributes zero)
#define HDIM   256     // hidden width per feature
#define KEFF   224     // ceil(TFLEN/32)*32 : effective K (a_n[d]=0 for d>=195)
#define NKSTEP 7       // KEFF / 32 (one WMMA K-step per chunk)
#define MT     64      // batch rows per workgroup
#define XSP    232     // xs pitch in bf16 elems (224 + 8 pad; 464B stride -> conflict-free b128)
#define WP     40      // W-chunk pitch in bf16 elems (32 + 8 pad; 80B stride -> conflict-free b128)
#define WELEM  (HDIM * WP)

typedef __bf16 bf16;
typedef __attribute__((ext_vector_type(16))) __bf16 v16bf;
typedef __attribute__((ext_vector_type(8)))  float  v8f;

union Frag { v16bf v; uint4 q[2]; };

__device__ __forceinline__ float elu1(float v) { return v > 0.f ? v : expm1f(v); }

__device__ __forceinline__ void load_frag_b(Frag& f, const bf16* p) {
  f.q[0] = *(const uint4*)(p);
  f.q[1] = *(const uint4*)(p + 16);
}

__global__ __launch_bounds__(256)
void fused_feat_mlp(const float* __restrict__ x,  const float* __restrict__ A,
                    const float* __restrict__ W1, const float* __restrict__ b1,
                    const float* __restrict__ W2, const float* __restrict__ b2,
                    float* __restrict__ out)
{
  extern __shared__ char smem[];
  bf16*  xs  = (bf16*)smem;                                  // [MT][XSP]
  bf16*  wb  = (bf16*)(smem + (size_t)MT * XSP * 2);         // [2][HDIM][WP] (transposed: [h][k])
  float* an  = (float*)(smem + (size_t)MT * XSP * 2 + 2ull * WELEM * 2); // [KEFF]
  float* red = an + KEFF;                                    // [MT]

  const int tid  = threadIdx.x;
  const int bx   = blockIdx.x;
  const int n    = bx >> 3;          // feature index; 8 adjacent blocks share W1[n] via L2
  const int mt   = bx & 7;           // which 64-row batch tile
  const int row0 = mt * MT;

  const int lane  = tid & 31;
  const int wave  = tid >> 5;
  const int wm    = wave & 3;        // 4 M-strips of 16 rows
  const int wn    = wave >> 2;       // 2 N-groups of 128 cols
  const int col   = lane & 15;       // WMMA N-column / A-row selector
  const int hi    = lane >> 4;       // hi half of wave
  const int hbase = wn * 128;

  // --- a_n = sigmoid(A[d, n]) masked (d < TF, d != n), zero-padded to KEFF ---
  if (tid < MT) red[tid] = 0.f;
  for (int d = tid; d < KEFF; d += 256) {
    float av = 0.f;
    if (d < TFLEN && d != n) av = 1.f / (1.f + expf(-A[d * DDIM + n]));
    an[d] = av;
  }
  __syncthreads();

  // --- stage scaled x tile as bf16: xs[r][d] = bf16(x[row0+r, d] * a_n[d]) ---
  for (int r = 0; r < MT; ++r) {
    if (tid < KEFF)
      xs[r * XSP + tid] = (bf16)(x[(size_t)(row0 + r) * DDIM + tid] * an[tid]);
  }
  // --- stage W1 chunk 0, transposed to [h][k] (coalesced global reads, h = tid) ---
  {
    const float* wsrc = W1 + ((size_t)n * DDIM) * HDIM + tid;
    bf16* wdst = wb;
    #pragma unroll
    for (int kc = 0; kc < 32; ++kc)
      wdst[tid * WP + kc] = (bf16)(wsrc[(size_t)kc * HDIM]);
  }
  __syncthreads();

  v8f acc[8];
  #pragma unroll
  for (int tt = 0; tt < 8; ++tt)
    #pragma unroll
    for (int i = 0; i < 8; ++i) acc[tt][i] = 0.f;

  // A-fragment source row for this wave (ISA 16-bit A 16x32 layout):
  // halves 0..7 = K + hi*8 + {0..7}, halves 8..15 = K + hi*8 + {16..23}
  const bf16* arowp = xs + (wm * 16 + col) * XSP + hi * 8;

  for (int c = 0; c < NKSTEP; ++c) {
    // L2 prefetch for the chunk two steps ahead (first-toucher block warms L2
    // for the 7 sibling blocks of this feature): 256 lanes x 128B = 32KB chunk
    if (c + 2 < NKSTEP) {
      const float* pf = W1 + ((size_t)n * DDIM + (size_t)(c + 2) * 32) * HDIM + tid * 32;
      __builtin_prefetch(pf, 0, 1);
    }
    // prefetch next W1 chunk into the other LDS buffer
    if (c + 1 < NKSTEP) {
      const float* wsrc = W1 + ((size_t)n * DDIM + (size_t)(c + 1) * 32) * HDIM + tid;
      bf16* wdst = wb + ((c + 1) & 1) * WELEM;
      #pragma unroll
      for (int kc = 0; kc < 32; ++kc)
        wdst[tid * WP + kc] = (bf16)(wsrc[(size_t)kc * HDIM]);
    }

    // A fragment for this K-step
    Frag a;
    const bf16* ap = arowp + c * 32;
    a.q[0] = *(const uint4*)(ap);
    a.q[1] = *(const uint4*)(ap + 16);

    // B (32x16, 16-bit): halves 0..7 = K hi*8 + {0..7}, halves 8..15 = K hi*8 + {16..23} (matches A).
    // Preload ALL 8 B fragments into distinct registers: the 16 ds_load_b128
    // issue as one batch, loads return in order, so WMMA tt only needs
    // dscnt <= 2*(7-tt) — the DS pipe drains underneath the matrix pipe
    // instead of a full s_wait_dscnt 0 before every WMMA.
    const bf16* bbase = wb + (c & 1) * WELEM + (size_t)(hbase + col) * WP + hi * 8;
    Frag bfrag[8];
    #pragma unroll
    for (int tt = 0; tt < 8; ++tt)
      load_frag_b(bfrag[tt], bbase + (size_t)tt * 16 * WP);

    #pragma unroll
    for (int tt = 0; tt < 8; ++tt)
      acc[tt] = __builtin_amdgcn_wmma_f32_16x16x32_bf16(
          false, a.v, false, bfrag[tt].v, (short)0, acc[tt], false, false);

    __syncthreads();
  }

  // --- fused layer 2: s[row] = sum_h elu(h1 + b1) * W2, reduced over h ---
  float s[8];
  #pragma unroll
  for (int r = 0; r < 8; ++r) s[r] = 0.f;

  #pragma unroll
  for (int tt = 0; tt < 8; ++tt) {
    const int h = hbase + tt * 16 + col;
    const float b1v = b1[(size_t)n * HDIM + h];
    const float w2v = W2[(size_t)n * HDIM + h];
    #pragma unroll
    for (int r = 0; r < 8; ++r) {
      float v = acc[tt][r] + b1v;
      s[r] += elu1(v) * w2v;
    }
  }
  // butterfly-reduce over the 16 h-columns held by this half-wave
  #pragma unroll
  for (int mask = 1; mask <= 8; mask <<= 1)
    #pragma unroll
    for (int r = 0; r < 8; ++r)
      s[r] += __shfl_xor(s[r], mask, 32);

  if (col == 0) {   // lanes 0 and 16: 8 distinct rows each
    #pragma unroll
    for (int r = 0; r < 8; ++r)
      atomicAdd(&red[wm * 16 + hi * 8 + r], s[r]);  // 2 N-waves add per row
  }
  __syncthreads();

  if (tid < MT) {
    float v = red[tid] + b2[n];
    out[(size_t)(row0 + tid) * DDIM + n] = elu1(v);
  }
}

extern "C" void kernel_launch(void* const* d_in, const int* in_sizes, int n_in,
                              void* d_out, int out_size, void* d_ws, size_t ws_size,
                              hipStream_t stream) {
  (void)in_sizes; (void)n_in; (void)out_size; (void)d_ws; (void)ws_size;
  const float* x  = (const float*)d_in[0];
  const float* A  = (const float*)d_in[1];
  const float* W1 = (const float*)d_in[2];
  const float* b1 = (const float*)d_in[3];
  const float* W2 = (const float*)d_in[4];
  const float* b2 = (const float*)d_in[5];
  float* out = (float*)d_out;

  constexpr size_t SMEM = (size_t)MT * XSP * 2      // xs tile
                        + 2ull * WELEM * 2          // double-buffered W chunk
                        + (size_t)KEFF * 4          // a_n
                        + (size_t)MT * 4;           // layer-2 reduction
  (void)hipFuncSetAttribute(reinterpret_cast<const void*>(fused_feat_mlp),
                            hipFuncAttributeMaxDynamicSharedMemorySize, (int)SMEM);
  fused_feat_mlp<<<dim3(DDIM * 8), dim3(256), SMEM, stream>>>(x, A, W1, b1, W2, b2, out);
}
